// VisionTransformer_80917183857253
// MI455X (gfx1250) — compile-verified, matrix operand order checked by reading, not run on hardware
//
#include <hip/hip_runtime.h>
#include <hip/hip_bf16.h>
#include <math.h>

// ---------------------------------------------------------------------------
// ViT forward for MI455X (gfx1250, wave32).
//  - All matmuls on v_wmma_f32_16x16x32_f16 (f16 A/B, f32 accumulate).
//  - RoPE fused into the Q/K projection epilogues via lane-pair shfl_xor
//    (the reference's dense [SEQ,128,128] rotary matrix is block-diagonal
//    2x2 rotations) -> saves ~79 GFLOP + a 77MB/layer memory pass.
//  - Weights converted+transposed to f16 once per call, so all weight-tile
//    staging is a straight contiguous copy.
//  - GEMM main loop is double-buffered: next tile staged with
//    GLOBAL_LOAD_ASYNC_TO_LDS_B128 (ASYNCcnt) while WMMAs run on the current
//    tile; s_wait_asynccnt only after compute. Sync fallback if the builtin
//    is not declared.
// ---------------------------------------------------------------------------

typedef __attribute__((ext_vector_type(16))) _Float16 v16h;
typedef __attribute__((ext_vector_type(8)))  float    v8f;

#if __has_builtin(__builtin_amdgcn_global_load_async_to_lds_b128)
#define VT_ASYNC 1
#else
#define VT_ASYNC 0
#endif

// The builtin's params are v4i pointers in explicit address spaces:
//   arg0: global (AS1) v4i*   arg1: LDS (AS3) v4i*   arg2: imm off  arg3: cpol
typedef int v4i_t __attribute__((vector_size(16)));
typedef __attribute__((address_space(1))) v4i_t GAS_v4i;
typedef __attribute__((address_space(3))) v4i_t LAS_v4i;

// 16-byte global -> LDS copy. Async (no VGPR round-trip, ASYNCcnt) when
// available; otherwise a plain synchronous 16B copy.
__device__ __forceinline__ void copy_b128_to_lds(const void* gsrc, void* ldst) {
#if VT_ASYNC
  __builtin_amdgcn_global_load_async_to_lds_b128((GAS_v4i*)gsrc, (LAS_v4i*)ldst,
                                                 0, 0);
#else
  *(uint4*)ldst = *(const uint4*)gsrc;
#endif
}

__device__ __forceinline__ void async_wait0() {
#if VT_ASYNC
#if __has_builtin(__builtin_amdgcn_s_wait_asynccnt)
  __builtin_amdgcn_s_wait_asynccnt(0);
#else
  asm volatile("s_wait_asynccnt 0x0" ::: "memory");
#endif
#endif
}

namespace cfg {
constexpr int B    = 64;
constexpr int C    = 14;
constexpr int SEQ  = 196;
constexpr int P    = 256;
constexpr int L    = 128;
constexpr int H    = 12;
constexpr int NL   = 4;
constexpr int HD   = H * L;        // 1536
constexpr int M    = B * SEQ;      // 12544 (divisible by 128)
constexpr int SEQP = 224;          // padded key length: 14 tiles of 16
constexpr int NQT  = 13;           // ceil(196/16) query tiles
constexpr int NKT  = SEQP / 16;    // 14 key tiles
constexpr int PVK  = SEQP / 32;    // 7 K-chunks for P@V
constexpr float SM_SCALE = 0.08838834764831845f;  // 128^-0.5
constexpr size_t ATT_LDS =
    (size_t)SEQP * L * sizeof(_Float16)     // K tile   [SEQP][L]
  + (size_t)L * SEQP * sizeof(_Float16)     // V^T tile [L][SEQP]
  + (size_t)16 * SEQP * sizeof(float)       // score rows [16][SEQP]
  + (size_t)16 * SEQP * sizeof(_Float16);   // prob rows  [16][SEQP] (f16)
}  // namespace cfg

__device__ __forceinline__ v8f wmma_f16f32(v16h a, v16h b, v8f c) {
  // D(16x16 f32) = A(16x32 f16) * B(32x16 f16) + C
  return __builtin_amdgcn_wmma_f32_16x16x32_f16(false, a, false, b, (short)0, c,
                                                false, false);
}

// Load a 16x32 A-fragment (or a 32x16 B-fragment stored column-major, i.e.
// tile[n][k] contiguous in k) from a tile with leading dimension `ld` halves.
// Layout per CDNA5 ISA 7.12.2 (16-bit A-matrix 16x32):
//   lanes 0-15 : row = lane,    v0..3 -> K=koff+0..7,  v4..7 -> K=koff+16..23
//   lanes16-31 : row = lane-16, v0..3 -> K=koff+8..15, v4..7 -> K=koff+24..31
__device__ __forceinline__ v16h frag_ld(const _Float16* tile, int row0, int ld,
                                        int koff) {
  const int lane = (int)(threadIdx.x & 31u);
  const _Float16* p =
      tile + (size_t)(row0 + (lane & 15)) * ld + koff + ((lane >> 4) << 3);
  v16h f;
#pragma unroll
  for (int i = 0; i < 4; ++i) {
    f[2 * i]         = p[2 * i];
    f[2 * i + 1]     = p[2 * i + 1];
    f[8 + 2 * i]     = p[16 + 2 * i];
    f[8 + 2 * i + 1] = p[16 + 2 * i + 1];
  }
  return f;
}

// Same fragment layout but loading Q rows straight from global memory
// (row stride HD halves), zero-filling padded query rows (s >= SEQ).
__device__ __forceinline__ v16h frag_ld_q(const _Float16* base, int s0,
                                          int koff) {
  const int lane = (int)(threadIdx.x & 31u);
  const int s = s0 + (lane & 15);
  v16h f;
#pragma unroll
  for (int i = 0; i < 16; ++i) f[i] = (_Float16)0.0f;
  if (s < cfg::SEQ) {
    const _Float16* p = base + (size_t)s * cfg::HD + koff + ((lane >> 4) << 3);
#pragma unroll
    for (int i = 0; i < 4; ++i) {
      f[2 * i]         = p[2 * i];
      f[2 * i + 1]     = p[2 * i + 1];
      f[8 + 2 * i]     = p[16 + 2 * i];
      f[8 + 2 * i + 1] = p[16 + 2 * i + 1];
    }
  }
  return f;
}

__device__ __forceinline__ float gelu_exact(float x) {
  return 0.5f * x * (1.0f + erff(x * 0.70710678118654752f));
}

// ---------------------------------------------------------------------------
// One-shot weight prep: convert fp32 W[K][N] to f16, pre-transposed Wt[N][K]
// so GEMM B-tiles become contiguous row copies (async-LDS eligible).
// ---------------------------------------------------------------------------
__global__ __launch_bounds__(256) void wtrans_kernel(
    const float* __restrict__ W, _Float16* __restrict__ Wt, int K, int N) {
  const int idx = (int)blockIdx.x * 256 + (int)threadIdx.x;
  if (idx >= K * N) return;
  const int k = idx / N;
  const int n = idx % N;
  Wt[(size_t)n * K + k] = (_Float16)W[idx];
}

// ---------------------------------------------------------------------------
// Generic tiled GEMM: out[M,N] = A[M,K] @ W[K,N] + bias (+GELU|+res|+RoPE)
// Wt is the pre-transposed f16 weight [N][K]. BM=BN=128, BK=32, 256 threads
// = 8 waves; each wave owns a 32x64 subtile (2x4 WMMA tiles).
// Double-buffered: tile t+1 is staged (async where it's a straight copy)
// before the WMMAs of tile t; ASYNCcnt is drained only after compute.
// EPI: 0 = bias, 1 = bias+GELU, 2 = bias+residual, 3 = bias+RoPE (Q/K proj)
// ---------------------------------------------------------------------------
template <bool AF16, bool OF16, int EPI>
__global__ __launch_bounds__(256) void gemm_kernel(
    const void* __restrict__ A_, const _Float16* __restrict__ Wt,
    const float* __restrict__ bias, const float* __restrict__ res,
    void* __restrict__ out_, int M, int N, int K) {
  constexpr int BM = 128, BN = 128, BK = 32;
  __shared__ _Float16 As[2][BM][BK];
  __shared__ _Float16 Bs[2][BN][BK];  // column-major of B: Bs[.][n][k]

  const int tid = (int)threadIdx.x;
  const int wave = tid >> 5;
  const int bm = (int)blockIdx.x * BM;
  const int bn = (int)blockIdx.y * BN;
  const int wr = (wave >> 1) * 32;  // wave row base within tile
  const int wc = (wave & 1) * 64;   // wave col base within tile
  // Per-thread staging coordinates.
  const int sr = tid >> 3;          // f32-A: row
  const int scol = (tid & 7) * 4;   //        col group of 4
  const int cr0 = (tid * 2) >> 2, cp0 = ((tid * 2) & 3) * 8;      // b128 #0
  const int cr1 = (tid * 2 + 1) >> 2, cp1 = ((tid * 2 + 1) & 3) * 8;  // #1

  v8f acc[2][4];
#pragma unroll
  for (int a = 0; a < 2; ++a)
#pragma unroll
    for (int b = 0; b < 4; ++b)
#pragma unroll
      for (int i = 0; i < 8; ++i) acc[a][b][i] = 0.0f;

  const int T = K / BK;

  // ---- prologue: stage tile 0 into buffer 0 ----
  if constexpr (!AF16) {
    const float* A = (const float*)A_;
#pragma unroll
    for (int p = 0; p < 4; ++p) {
      const int r = p * 32 + sr;
      const float4 v = *(const float4*)&A[(size_t)(bm + r) * K + scol];
      As[0][r][scol + 0] = (_Float16)v.x;
      As[0][r][scol + 1] = (_Float16)v.y;
      As[0][r][scol + 2] = (_Float16)v.z;
      As[0][r][scol + 3] = (_Float16)v.w;
    }
  } else {
    const _Float16* A = (const _Float16*)A_;
    copy_b128_to_lds(&A[(size_t)(bm + cr0) * K + cp0], &As[0][cr0][cp0]);
    copy_b128_to_lds(&A[(size_t)(bm + cr1) * K + cp1], &As[0][cr1][cp1]);
  }
  copy_b128_to_lds(&Wt[(size_t)(bn + cr0) * K + cp0], &Bs[0][cr0][cp0]);
  copy_b128_to_lds(&Wt[(size_t)(bn + cr1) * K + cp1], &Bs[0][cr1][cp1]);
  async_wait0();
  __syncthreads();

  for (int t = 0; t < T; ++t) {
    const int cur = t & 1, nxt = cur ^ 1;
    const bool have_next = (t + 1 < T);
    const int k1 = (t + 1) * BK;

    // ---- issue staging for tile t+1 (overlaps with this tile's WMMAs) ----
    float4 areg[4];
    if (have_next) {
      copy_b128_to_lds(&Wt[(size_t)(bn + cr0) * K + k1 + cp0],
                       &Bs[nxt][cr0][cp0]);
      copy_b128_to_lds(&Wt[(size_t)(bn + cr1) * K + k1 + cp1],
                       &Bs[nxt][cr1][cp1]);
      if constexpr (AF16) {
        const _Float16* A = (const _Float16*)A_;
        copy_b128_to_lds(&A[(size_t)(bm + cr0) * K + k1 + cp0],
                         &As[nxt][cr0][cp0]);
        copy_b128_to_lds(&A[(size_t)(bm + cr1) * K + k1 + cp1],
                         &As[nxt][cr1][cp1]);
      } else {
        const float* A = (const float*)A_;
#pragma unroll
        for (int p = 0; p < 4; ++p)
          areg[p] = *(const float4*)&A[(size_t)(bm + p * 32 + sr) * K + k1 +
                                       scol];
      }
    }

    // ---- 8 WMMAs on the current tile ----
    const v16h af0 = frag_ld(&As[cur][0][0], wr, BK, 0);
    const v16h af1 = frag_ld(&As[cur][0][0], wr + 16, BK, 0);
    v16h bf[4];
#pragma unroll
    for (int ct = 0; ct < 4; ++ct)
      bf[ct] = frag_ld(&Bs[cur][0][0], wc + ct * 16, BK, 0);
#pragma unroll
    for (int ct = 0; ct < 4; ++ct) {
      acc[0][ct] = wmma_f16f32(af0, bf[ct], acc[0][ct]);
      acc[1][ct] = wmma_f16f32(af1, bf[ct], acc[1][ct]);
    }

    // ---- deferred fp32->f16 convert + store of the next A tile ----
    if constexpr (!AF16) {
      if (have_next) {
#pragma unroll
        for (int p = 0; p < 4; ++p) {
          const int r = p * 32 + sr;
          As[nxt][r][scol + 0] = (_Float16)areg[p].x;
          As[nxt][r][scol + 1] = (_Float16)areg[p].y;
          As[nxt][r][scol + 2] = (_Float16)areg[p].z;
          As[nxt][r][scol + 3] = (_Float16)areg[p].w;
        }
      }
    }
    async_wait0();
    __syncthreads();
  }

  // ---- epilogue; C/D layout: VGPR i -> row (lane<16 ? i : i+8), col lane&15
  const int lane = tid & 31;
  const int rsel = (lane >> 4) * 8;
  const int csel = lane & 15;
#pragma unroll
  for (int rt = 0; rt < 2; ++rt) {
#pragma unroll
    for (int ct = 0; ct < 4; ++ct) {
#pragma unroll
      for (int i = 0; i < 8; ++i) {
        const int row = bm + wr + rt * 16 + rsel + i;
        const int col = bn + wc + ct * 16 + csel;
        float v = acc[rt][ct][i] + bias[col];
        if constexpr (EPI == 1) v = gelu_exact(v);
        if constexpr (EPI == 2) v += res[(size_t)row * N + col];
        if constexpr (EPI == 3) {
          // Fused analytic 2D RoPE. Pair partner (col^1) sits in lane^1.
          // First half (d<64): theta = (c%14)*1000^(-2j/64); else uses c/14.
          const int cpos = row % cfg::SEQ;
          const int d = col & (cfg::L - 1);
          const int j = (d < 64) ? (d >> 1) : ((d - 64) >> 1);
          const int pos = (d < 64) ? (cpos % cfg::C) : (cpos / cfg::C);
          const float freq = powf(1000.0f, -2.0f * (float)j / 64.0f);
          const float th = (float)pos * freq;
          const float cs = cosf(th), sn = sinf(th);
          const float partner = __shfl_xor(v, 1, 32);
          v = ((col & 1) == 0) ? (cs * v + sn * partner)
                               : (cs * v - sn * partner);
        }
        if constexpr (OF16)
          ((_Float16*)out_)[(size_t)row * N + col] = (_Float16)v;
        else
          ((float*)out_)[(size_t)row * N + col] = v;
      }
    }
  }
}

// ---------------------------------------------------------------------------
// RMSNorm: one wave per 128-wide row; lane holds 4 floats; shfl_xor reduce.
// ---------------------------------------------------------------------------
__global__ __launch_bounds__(128) void rmsnorm_kernel(
    const float* __restrict__ x, const float* __restrict__ scale,
    float* __restrict__ out) {
  const int wave = (int)(threadIdx.x >> 5);
  const int lane = (int)(threadIdx.x & 31u);
  const size_t row = (size_t)blockIdx.x * 4 + wave;
  const float4 v = *(const float4*)&x[row * cfg::L + lane * 4];
  float ss = v.x * v.x + v.y * v.y + v.z * v.z + v.w * v.w;
#pragma unroll
  for (int m = 16; m >= 1; m >>= 1) ss += __shfl_xor(ss, m, 32);
  const float inv = rsqrtf(ss * (1.0f / cfg::L) + 1e-6f);
  const float4 sc = *(const float4*)&scale[lane * 4];
  float4 o;
  o.x = v.x * inv * sc.x;
  o.y = v.y * inv * sc.y;
  o.z = v.z * inv * sc.z;
  o.w = v.w * inv * sc.w;
  *(float4*)&out[row * cfg::L + lane * 4] = o;
}

// ---------------------------------------------------------------------------
// Attention: one block (8 waves) per (b,h). K staged row-major via async-LDS
// copies; V staged transposed (element scatter). 13 query tiles of 16 rows,
// WMMA for S=QK^T and O=PV, LDS softmax in between (padded keys -> p=0).
// `att` may alias `rq`: reads of tile t complete before tile t's writes.
// ---------------------------------------------------------------------------
__global__ __launch_bounds__(256) void attn_kernel(
    const _Float16* __restrict__ rq, const _Float16* __restrict__ rk,
    const _Float16* __restrict__ vv, _Float16* __restrict__ att) {
  using namespace cfg;
  extern __shared__ char smem[];
  _Float16* Kt  = (_Float16*)smem;                    // [SEQP][L]
  _Float16* VtT = Kt + (size_t)SEQP * L;              // [L][SEQP]
  float*    Sc  = (float*)(VtT + (size_t)L * SEQP);   // [16][SEQP]
  _Float16* Pf  = (_Float16*)(Sc + 16 * SEQP);        // [16][SEQP]

  const int tid = (int)threadIdx.x;
  const int wave = tid >> 5;
  const int lane = tid & 31;
  const int b = (int)blockIdx.x / H;
  const int h = (int)blockIdx.x % H;
  const _Float16* qbase = rq + (size_t)b * SEQ * HD + (size_t)h * L;
  const _Float16* kbase = rk + (size_t)b * SEQ * HD + (size_t)h * L;
  const _Float16* vbase = vv + (size_t)b * SEQ * HD + (size_t)h * L;
  _Float16* obase = att + (size_t)b * SEQ * HD + (size_t)h * L;

  // Zero the padded rows of Kt / padded columns of VtT (28 x 128 each).
  for (int i = tid; i < (SEQP - SEQ) * L; i += 256) {
    const int s = SEQ + i / L, d = i % L;
    Kt[s * L + d] = (_Float16)0.0f;
    VtT[d * SEQP + s] = (_Float16)0.0f;
  }
  // K rows: straight copy -> async b128 (196 rows x 16 chunks = 3136).
  for (int c = tid; c < SEQ * (L / 8); c += 256) {
    const int s = c >> 4, part = (c & 15) * 8;
    copy_b128_to_lds(&kbase[(size_t)s * HD + part], &Kt[s * L + part]);
  }
  // V transposed: element scatter (not async-able at useful granularity).
  for (int i = tid; i < SEQ * L; i += 256) {
    const int s = i / L, d = i % L;
    VtT[d * SEQP + s] = vbase[(size_t)s * HD + d];
  }
  async_wait0();
  __syncthreads();

  for (int qt = 0; qt < NQT; ++qt) {
    // Preload the 4 K-chunk fragments of this 16-row query tile.
    v16h aq[4];
#pragma unroll
    for (int kc = 0; kc < 4; ++kc) aq[kc] = frag_ld_q(qbase, qt * 16, kc * 32);

    // S = Q K^T : 14 key tiles distributed over 8 waves.
    for (int kt = wave; kt < NKT; kt += 8) {
      v8f s8 = {0.f, 0.f, 0.f, 0.f, 0.f, 0.f, 0.f, 0.f};
#pragma unroll
      for (int kc = 0; kc < 4; ++kc) {
        const v16h bk = frag_ld(Kt, kt * 16, L, kc * 32);
        s8 = wmma_f16f32(aq[kc], bk, s8);
      }
      const int mrow = (lane >> 4) * 8;
      const int ncol = kt * 16 + (lane & 15);
#pragma unroll
      for (int i = 0; i < 8; ++i)
        Sc[(mrow + i) * SEQP + ncol] = s8[i] * SM_SCALE;
    }
    __syncthreads();

    // Row-wise softmax over the 196 valid keys; padded keys get p = 0.
    {
      const int mr = tid >> 4;   // 0..15
      const int l16 = tid & 15;  // 16 lanes per row (16-aligned within wave)
      float mx = -1e30f;
      for (int n = l16; n < SEQ; n += 16) mx = fmaxf(mx, Sc[mr * SEQP + n]);
#pragma unroll
      for (int o = 8; o >= 1; o >>= 1) mx = fmaxf(mx, __shfl_xor(mx, o, 32));
      float sum = 0.0f;
      for (int n = l16; n < SEQ; n += 16) sum += expf(Sc[mr * SEQP + n] - mx);
#pragma unroll
      for (int o = 8; o >= 1; o >>= 1) sum += __shfl_xor(sum, o, 32);
      const float inv = 1.0f / sum;
      for (int n = l16; n < SEQP; n += 16) {
        const float p = (n < SEQ) ? expf(Sc[mr * SEQP + n] - mx) * inv : 0.0f;
        Pf[mr * SEQP + n] = (_Float16)p;
      }
    }
    __syncthreads();

    // O = P V : each wave owns a 16-column strip of the 128-wide output.
    {
      v8f o8 = {0.f, 0.f, 0.f, 0.f, 0.f, 0.f, 0.f, 0.f};
      for (int kc = 0; kc < PVK; ++kc) {
        const v16h ap = frag_ld(Pf, 0, SEQP, kc * 32);
        const v16h bv = frag_ld(VtT, wave * 16, SEQP, kc * 32);
        o8 = wmma_f16f32(ap, bv, o8);
      }
      const int mrow = (lane >> 4) * 8;
      const int ncol = wave * 16 + (lane & 15);
#pragma unroll
      for (int i = 0; i < 8; ++i) {
        const int s = qt * 16 + mrow + i;
        if (s < SEQ) obase[(size_t)s * HD + ncol] = (_Float16)o8[i];
      }
    }
    __syncthreads();
  }
}

// ---------------------------------------------------------------------------
// Orchestration. Inputs (setup_inputs insertion order):
//   0:x  1:embed_W  2:embed_b
//   3 + layer*14 + {0:Wq 1:bq 2:Wk 3:bk 4:Wv 5:bv 6:Wo 7:bo 8:W1 9:b1 10:W2
//                   11:b2 12:s1 13:s2}
//   59:r (unused — RoPE computed analytically, fused into Q/K projections)
// Output: final h [B,SEQ,L] f32 in d_out (also used as the running h buffer).
// ---------------------------------------------------------------------------
extern "C" void kernel_launch(void* const* d_in, const int* in_sizes, int n_in,
                              void* d_out, int out_size, void* d_ws,
                              size_t ws_size, hipStream_t stream) {
  (void)in_sizes; (void)n_in; (void)out_size; (void)ws_size;
  using namespace cfg;

  const float* x  = (const float*)d_in[0];
  const float* eW = (const float*)d_in[1];
  const float* eb = (const float*)d_in[2];
  float* h = (float*)d_out;

  char* ws = (char*)d_ws;
  size_t off = 0;
  float* nx  = (float*)(ws + off); off += (size_t)M * L * sizeof(float);
  float* mid = (float*)(ws + off); off += (size_t)M * L * sizeof(float);
  _Float16* q = (_Float16*)(ws + off); off += (size_t)M * HD * sizeof(_Float16);
  _Float16* k = (_Float16*)(ws + off); off += (size_t)M * HD * sizeof(_Float16);
  _Float16* v = (_Float16*)(ws + off); off += (size_t)M * HD * sizeof(_Float16);
  // Pre-transposed f16 weights.
  _Float16* eWt = (_Float16*)(ws + off); off += (size_t)L * P * sizeof(_Float16);
  _Float16* WT[NL][6];  // 0:WqT 1:WkT 2:WvT [HD][L]; 3:WoT [L][HD]; 4:W1T 5:W2T
  for (int l = 0; l < NL; ++l) {
    for (int wdx = 0; wdx < 3; ++wdx) {
      WT[l][wdx] = (_Float16*)(ws + off);
      off += (size_t)HD * L * sizeof(_Float16);
    }
    WT[l][3] = (_Float16*)(ws + off); off += (size_t)L * HD * sizeof(_Float16);
    WT[l][4] = (_Float16*)(ws + off); off += (size_t)L * L * sizeof(_Float16);
    WT[l][5] = (_Float16*)(ws + off); off += (size_t)L * L * sizeof(_Float16);
  }

  const dim3 blk(256);
  const dim3 gEmb(M / 128, 1);
  const dim3 gQKV(M / 128, HD / 128);
  const dim3 gSq(M / 128, 1);

  // ---- one-shot weight convert + transpose (f32 [K][N] -> f16 [N][K]) ----
  wtrans_kernel<<<(P * L + 255) / 256, blk, 0, stream>>>(eW, eWt, P, L);
  for (int l = 0; l < NL; ++l) {
    const float* Wq = (const float*)d_in[3 + l * 14 + 0];
    const float* Wk = (const float*)d_in[3 + l * 14 + 2];
    const float* Wv = (const float*)d_in[3 + l * 14 + 4];
    const float* Wo = (const float*)d_in[3 + l * 14 + 6];
    const float* W1 = (const float*)d_in[3 + l * 14 + 8];
    const float* W2 = (const float*)d_in[3 + l * 14 + 10];
    wtrans_kernel<<<(L * HD + 255) / 256, blk, 0, stream>>>(Wq, WT[l][0], L, HD);
    wtrans_kernel<<<(L * HD + 255) / 256, blk, 0, stream>>>(Wk, WT[l][1], L, HD);
    wtrans_kernel<<<(L * HD + 255) / 256, blk, 0, stream>>>(Wv, WT[l][2], L, HD);
    wtrans_kernel<<<(HD * L + 255) / 256, blk, 0, stream>>>(Wo, WT[l][3], HD, L);
    wtrans_kernel<<<(L * L + 255) / 256, blk, 0, stream>>>(W1, WT[l][4], L, L);
    wtrans_kernel<<<(L * L + 255) / 256, blk, 0, stream>>>(W2, WT[l][5], L, L);
  }

  // h = x @ embed_W + embed_b
  gemm_kernel<false, false, 0><<<gEmb, blk, 0, stream>>>(
      x, eWt, eb, nullptr, h, M, L, P);

  for (int l = 0; l < NL; ++l) {
    const float* bq = (const float*)d_in[3 + l * 14 + 1];
    const float* bk = (const float*)d_in[3 + l * 14 + 3];
    const float* bv = (const float*)d_in[3 + l * 14 + 5];
    const float* bo = (const float*)d_in[3 + l * 14 + 7];
    const float* b1 = (const float*)d_in[3 + l * 14 + 9];
    const float* b2 = (const float*)d_in[3 + l * 14 + 11];
    const float* s1 = (const float*)d_in[3 + l * 14 + 12];
    const float* s2 = (const float*)d_in[3 + l * 14 + 13];

    // nx = rmsnorm(h, s1)
    rmsnorm_kernel<<<M / 4, 128, 0, stream>>>(h, s1, nx);
    // q/k = RoPE(nx @ W* + b*), v = nx @ Wv + bv  (f16 outputs for WMMA attn)
    gemm_kernel<false, true, 3><<<gQKV, blk, 0, stream>>>(
        nx, WT[l][0], bq, nullptr, q, M, HD, L);
    gemm_kernel<false, true, 3><<<gQKV, blk, 0, stream>>>(
        nx, WT[l][1], bk, nullptr, k, M, HD, L);
    gemm_kernel<false, true, 0><<<gQKV, blk, 0, stream>>>(
        nx, WT[l][2], bv, nullptr, v, M, HD, L);
    // Attention; output aliases the q buffer (safe: per-tile read-then-write).
    attn_kernel<<<B * H, 256, (unsigned)ATT_LDS, stream>>>(q, k, v, q);
    // h = h + att @ Wo + bo   (A is f16, residual epilogue, in-place on h)
    gemm_kernel<true, false, 2><<<gSq, blk, 0, stream>>>(
        q, WT[l][3], bo, h, h, M, L, HD);
    // MLP: nx = rmsnorm(h, s2); mid = gelu(nx@W1+b1); h = h + mid@W2+b2
    rmsnorm_kernel<<<M / 4, 128, 0, stream>>>(h, s2, nx);
    gemm_kernel<false, false, 1><<<gSq, blk, 0, stream>>>(
        nx, WT[l][4], b1, nullptr, mid, M, L, L);
    gemm_kernel<false, false, 2><<<gSq, blk, 0, stream>>>(
        mid, WT[l][5], b2, h, h, M, L, L);
  }
}
